// Attention_62277025792152
// MI455X (gfx1250) — matrix operand order repaired and checked by reading, not compiled, not run on hardware
//
#include <hip/hip_runtime.h>
#include <hip/hip_bf16.h>
#include <math.h>

// ---------------------------------------------------------------------------
// Types for CDNA5 WMMA (wave32, 16x16x32 bf16 -> f32)
// ---------------------------------------------------------------------------
typedef __bf16 bf16_t;
typedef __attribute__((ext_vector_type(8)))  bf16_t bf16x8;
typedef __attribute__((ext_vector_type(16))) bf16_t v16bf;
typedef __attribute__((ext_vector_type(8)))  float  v8f;
typedef __attribute__((ext_vector_type(4)))  unsigned int v4u;
typedef __attribute__((ext_vector_type(8)))  int v8i;
typedef __attribute__((ext_vector_type(4)))  int v4i;

#define SEQ 4096
#define HID 1024
#define BM 128          // block tile M
#define BN 128          // block tile N
#define BK 64           // K step per staged tile
#define LP 72           // padded LDS row pitch in bf16 (144 B, 16B-aligned)

#if __has_builtin(__builtin_amdgcn_tensor_load_to_lds) && __has_builtin(__builtin_amdgcn_s_wait_tensorcnt)
#define HAVE_TDM 1
#if __has_include(<hip/amd_detail/amd_gfx1250_TDM.h>)
#define TDM_6ARG 1
#endif
#endif

// Compile-time path probes (visible in compiler stderr).
#if defined(HAVE_TDM)
#if defined(TDM_6ARG)
#pragma message("CDNA5 probe: TDM path = builtin 6-arg form")
#else
#pragma message("CDNA5 probe: TDM path = builtin 5-arg form")
#endif
#else
#pragma message("CDNA5 probe: TDM builtin NOT available -> manual staging fallback")
#endif

// Assemble a 16-element bf16 fragment from two contiguous 16-byte LDS chunks.
static __device__ __forceinline__ v16bf ld_frag(const bf16_t* p0, const bf16_t* p1) {
    bf16x8 lo = *reinterpret_cast<const bf16x8*>(p0);
    bf16x8 hi = *reinterpret_cast<const bf16x8*>(p1);
    return __builtin_shufflevector(lo, hi, 0,1,2,3,4,5,6,7,8,9,10,11,12,13,14,15);
}

#if defined(HAVE_TDM)
// TDM 2D tile load: (tile_h rows x tile_w elems, 2-byte elems) from a
// row-major tensor (width tensor_w, height tensor_h, row stride `stride`
// elems) into LDS at byte offset lds_off; pads each 32-DWORD row chunk with
// 4 DWORDs -> LDS pitch 72 bf16. Descriptor per CDNA5 ISA.
static __device__ __forceinline__ void tdm_load_2d(
    unsigned lds_off, const bf16_t* gptr,
    unsigned tensor_w, unsigned tensor_h,
    unsigned tile_w, unsigned tile_h, unsigned stride)
{
    unsigned long long ga = (unsigned long long)(uintptr_t)gptr;
    v4u g0;
    g0[0] = 1u;                                   // count=1 (valid), user mode
    g0[1] = lds_off;                              // lds_addr (bytes)
    g0[2] = (unsigned)(ga & 0xFFFFFFFFu);         // global_addr[31:0]
    g0[3] = (unsigned)((ga >> 32) & 0x01FFFFFFu)  // global_addr[56:32]
          | (2u << 30);                           // type = 2 (image)
    v8i g1;
    g1[0] = (int)((1u << 16)                      // data_size = 1 -> 2 bytes
                | (1u << 20)                      // pad_enable
                | (4u << 22)                      // pad_interval: 32 DWORDs
                | (3u << 25));                    // pad_amount: 4 DWORDs
    g1[1] = (int)((tensor_w & 0xFFFFu) << 16);                                  // dim0[15:0]
    g1[2] = (int)(((tensor_w >> 16) & 0xFFFFu) | ((tensor_h & 0xFFFFu) << 16)); // dim0[31:16]|dim1[15:0]
    g1[3] = (int)(((tensor_h >> 16) & 0xFFFFu) | ((tile_w & 0xFFFFu) << 16));   // dim1[31:16]|tile_dim0
    g1[4] = (int)(tile_h & 0xFFFFu);              // tile_dim1 (tile_dim2 = 0)
    g1[5] = (int)stride;                          // tensor_dim0_stride[31:0]
    g1[6] = 0;
    g1[7] = 0;
    v4i z4 = {0, 0, 0, 0};
#if defined(TDM_6ARG)
    v8i z8 = {0, 0, 0, 0, 0, 0, 0, 0};
    __builtin_amdgcn_tensor_load_to_lds(g0, g1, z4, z4, z8, 0);
#else
    __builtin_amdgcn_tensor_load_to_lds(g0, g1, z4, z4, 0);
#endif
}
#endif

static __device__ __forceinline__ unsigned lds_off_of(const void* p) {
    return (unsigned)(uintptr_t)p;
}

// ---------------------------------------------------------------------------
// Wave compute core: 8 waves as 4(M) x 2(N); wave tile 32x64; 16 WMMA / tile.
// All 6 fragments of a K-half are preloaded into distinct registers before
// the 8 WMMAs issue, so ds_load latency is covered by one graded wait and the
// matrix ops run back-to-back.
// ---------------------------------------------------------------------------
static __device__ __forceinline__ void wave_mma(
    const bf16_t (*As)[LP], const bf16_t (*Bt)[LP],
    int waveM, int waveN, int lrow, int kh, v8f (&acc)[2][4])
{
    #pragma unroll
    for (int kk = 0; kk < BK; kk += 32) {
        v16bf af[2], bfr[4];
        #pragma unroll
        for (int mt = 0; mt < 2; ++mt) {
            const int r = waveM * 32 + mt * 16 + lrow;
            af[mt] = ld_frag(&As[r][kk + kh * 8], &As[r][kk + 16 + kh * 8]);
        }
        #pragma unroll
        for (int nt = 0; nt < 4; ++nt) {
            const int c = waveN * 64 + nt * 16 + lrow;
            bfr[nt] = ld_frag(&Bt[c][kk + kh * 8], &Bt[c][kk + 16 + kh * 8]);
        }
        #pragma unroll
        for (int nt = 0; nt < 4; ++nt) {
            acc[0][nt] = __builtin_amdgcn_wmma_f32_16x16x32_bf16(
                false, af[0], false, bfr[nt], (short)0, acc[0][nt], false, false);
            acc[1][nt] = __builtin_amdgcn_wmma_f32_16x16x32_bf16(
                false, af[1], false, bfr[nt], (short)0, acc[1][nt], false, false);
        }
    }
}

#define GEMM_WAVE_SETUP()                                            \
    const int tid  = threadIdx.x;                                    \
    const int wave = tid >> 5;                                       \
    const int lane = tid & 31;                                       \
    const int waveM = wave & 3;                                      \
    const int waveN = wave >> 2;                                     \
    const int lrow  = lane & 15;                                     \
    const int kh    = lane >> 4;                                     \
    (void)lane;                                                      \
    v8f acc[2][4] = {{v8f{}, v8f{}, v8f{}, v8f{}},                   \
                     {v8f{}, v8f{}, v8f{}, v8f{}}};

// ---------------------------------------------------------------------------
// Projection: P = cvt_bf16( X(f32)[4096,1024] @ W(f32)[1024,1024] + b )
// Double-buffered manual staging (f32 -> bf16 on the way into LDS).
// ---------------------------------------------------------------------------
static __device__ __forceinline__ void stage_proj(
    bf16_t (*As)[LP], bf16_t (*Bt)[LP],
    const float* __restrict__ X, const float* __restrict__ W,
    int m0, int n0, int k0, int tid)
{
    const int rA = tid >> 1;           // 0..127
    const int cA = (tid & 1) * 32;     // 0 / 32
    const float* xa = X + (size_t)(m0 + rA) * HID + k0 + cA;
    #pragma unroll
    for (int j = 0; j < 4; ++j) {
        float4 f0 = *reinterpret_cast<const float4*>(xa + j * 8 + 0);
        float4 f1 = *reinterpret_cast<const float4*>(xa + j * 8 + 4);
        bf16x8 o;
        o[0]=(bf16_t)f0.x; o[1]=(bf16_t)f0.y; o[2]=(bf16_t)f0.z; o[3]=(bf16_t)f0.w;
        o[4]=(bf16_t)f1.x; o[5]=(bf16_t)f1.y; o[6]=(bf16_t)f1.z; o[7]=(bf16_t)f1.w;
        *reinterpret_cast<bf16x8*>(&As[rA][cA + j * 8]) = o;
    }
    const int rB = tid >> 2;           // k 0..63
    const int cB = (tid & 3) * 32;     // n 0..96
    const float* wb = W + (size_t)(k0 + rB) * HID + n0 + cB;
    #pragma unroll
    for (int j = 0; j < 8; ++j) {
        float4 f = *reinterpret_cast<const float4*>(wb + j * 4);
        Bt[cB + j * 4 + 0][rB] = (bf16_t)f.x;
        Bt[cB + j * 4 + 1][rB] = (bf16_t)f.y;
        Bt[cB + j * 4 + 2][rB] = (bf16_t)f.z;
        Bt[cB + j * 4 + 3][rB] = (bf16_t)f.w;
    }
}

__global__ __launch_bounds__(256) void proj_kernel(
    const float* __restrict__ X, const float* __restrict__ W,
    const float* __restrict__ bias, bf16_t* __restrict__ P)
{
    __shared__ bf16_t As[2][BM][LP];
    __shared__ bf16_t Bt[2][BN][LP];
    GEMM_WAVE_SETUP();
    const int m0 = blockIdx.x * BM;
    const int n0 = blockIdx.y * BN;

    stage_proj(As[0], Bt[0], X, W, m0, n0, 0, tid);
    __syncthreads();

    const int NIT = HID / BK;
    for (int it = 0; it < NIT; ++it) {
        const int cur = it & 1;
        if (it + 1 < NIT)
            stage_proj(As[cur ^ 1], Bt[cur ^ 1], X, W, m0, n0, (it + 1) * BK, tid);
        wave_mma(As[cur], Bt[cur], waveM, waveN, lrow, kh, acc);
        __syncthreads();
    }

    #pragma unroll
    for (int mt = 0; mt < 2; ++mt) {
        const int mBase = m0 + waveM * 32 + mt * 16 + kh * 8;
        #pragma unroll
        for (int nt = 0; nt < 4; ++nt) {
            const int n = n0 + waveN * 64 + nt * 16 + lrow;
            const float bv = bias[n];
            #pragma unroll
            for (int i = 0; i < 8; ++i)
                P[(size_t)(mBase + i) * HID + n] = (bf16_t)(acc[mt][nt][i] + bv);
        }
    }
}

// ---------------------------------------------------------------------------
// Scores: S(f32)[4096,4096] = (Qb @ Kb^T) / 32.
// Both tiles are straight 2D bf16 copies -> TDM double-buffered async stage:
// wave 0 issues tile k+1, all waves compute tile k, TENSORcnt drained before
// the single block barrier.
// ---------------------------------------------------------------------------
static __device__ __forceinline__ void stage_scores_manual(
    bf16_t (*As)[LP], bf16_t (*Bt)[LP],
    const bf16_t* __restrict__ Qb, const bf16_t* __restrict__ Kb,
    int m0, int n0, int k0, int tid)
{
    const int r = tid >> 1, c = (tid & 1) * 32;
    const bf16_t* qa = Qb + (size_t)(m0 + r) * HID + k0 + c;
    const bf16_t* kb = Kb + (size_t)(n0 + r) * HID + k0 + c;
    #pragma unroll
    for (int j = 0; j < 4; ++j) {
        *reinterpret_cast<int4*>(&As[r][c + j * 8]) =
            *reinterpret_cast<const int4*>(qa + j * 8);
        *reinterpret_cast<int4*>(&Bt[r][c + j * 8]) =
            *reinterpret_cast<const int4*>(kb + j * 8);
    }
}

__global__ __launch_bounds__(256) void scores_kernel(
    const bf16_t* __restrict__ Qb, const bf16_t* __restrict__ Kb,
    float* __restrict__ S)
{
    __shared__ bf16_t As[2][BM][LP];
    __shared__ bf16_t Bt[2][BN][LP];
    GEMM_WAVE_SETUP();
    const int m0 = blockIdx.x * BM;
    const int n0 = blockIdx.y * BN;
    const int NIT = HID / BK;

#if defined(HAVE_TDM)
    if (wave == 0) {
        tdm_load_2d(lds_off_of(&As[0][0][0]), Qb + (size_t)m0 * HID,
                    HID, SEQ, BK, BM, HID);
        tdm_load_2d(lds_off_of(&Bt[0][0][0]), Kb + (size_t)n0 * HID,
                    HID, SEQ, BK, BN, HID);
        __builtin_amdgcn_s_wait_tensorcnt(0);
    }
#else
    stage_scores_manual(As[0], Bt[0], Qb, Kb, m0, n0, 0, tid);
#endif
    __syncthreads();

    for (int it = 0; it < NIT; ++it) {
        const int cur = it & 1;
#if defined(HAVE_TDM)
        if (wave == 0 && it + 1 < NIT) {
            const int kn = (it + 1) * BK;
            tdm_load_2d(lds_off_of(&As[cur ^ 1][0][0]), Qb + (size_t)m0 * HID + kn,
                        HID, SEQ, BK, BM, HID);
            tdm_load_2d(lds_off_of(&Bt[cur ^ 1][0][0]), Kb + (size_t)n0 * HID + kn,
                        HID, SEQ, BK, BN, HID);
        }
#else
        if (it + 1 < NIT)
            stage_scores_manual(As[cur ^ 1], Bt[cur ^ 1], Qb, Kb, m0, n0,
                                (it + 1) * BK, tid);
#endif
        wave_mma(As[cur], Bt[cur], waveM, waveN, lrow, kh, acc);
#if defined(HAVE_TDM)
        if (wave == 0) __builtin_amdgcn_s_wait_tensorcnt(0);
#endif
        __syncthreads();
    }

    const float scale = 0.03125f;  // 1/sqrt(1024)
    #pragma unroll
    for (int mt = 0; mt < 2; ++mt) {
        const int mBase = m0 + waveM * 32 + mt * 16 + kh * 8;
        #pragma unroll
        for (int nt = 0; nt < 4; ++nt) {
            const int n = n0 + waveN * 64 + nt * 16 + lrow;
            #pragma unroll
            for (int i = 0; i < 8; ++i)
                S[(size_t)(mBase + i) * SEQ + n] = acc[mt][nt][i] * scale;
        }
    }
}

// ---------------------------------------------------------------------------
// Row softmax, in place: read f32 row, write bf16 probabilities over the same
// row start (row pitch stays 4096 f32 = 8192 bf16 slots). One block per row.
// ---------------------------------------------------------------------------
__global__ __launch_bounds__(256) void softmax_kernel(float* __restrict__ S)
{
    const int row = blockIdx.x;
    const int tid = threadIdx.x;
    float* rp = S + (size_t)row * SEQ;

    float v[16];
    float m = -3.402823466e+38f;
    #pragma unroll
    for (int i = 0; i < 16; ++i) { v[i] = rp[tid + i * 256]; m = fmaxf(m, v[i]); }

    #pragma unroll
    for (int off = 16; off > 0; off >>= 1) m = fmaxf(m, __shfl_xor(m, off, 32));

    __shared__ float red[8];
    if ((tid & 31) == 0) red[tid >> 5] = m;
    __syncthreads();
    float rowMax = red[0];
    #pragma unroll
    for (int w = 1; w < 8; ++w) rowMax = fmaxf(rowMax, red[w]);
    __syncthreads();

    float s = 0.0f;
    float e[16];
    #pragma unroll
    for (int i = 0; i < 16; ++i) { e[i] = __expf(v[i] - rowMax); s += e[i]; }
    #pragma unroll
    for (int off = 16; off > 0; off >>= 1) s += __shfl_xor(s, off, 32);
    if ((tid & 31) == 0) red[tid >> 5] = s;
    __syncthreads();
    float rowSum = 0.0f;
    #pragma unroll
    for (int w = 0; w < 8; ++w) rowSum += red[w];
    const float inv = 1.0f / rowSum;

    // All f32 reads complete before the barriers above -> safe in-place bf16.
    bf16_t* op = reinterpret_cast<bf16_t*>(rp);
    #pragma unroll
    for (int i = 0; i < 16; ++i) op[tid + i * 256] = (bf16_t)(e[i] * inv);
}

// ---------------------------------------------------------------------------
// Context: Out(f32)[4096,1024] = P(bf16, pitch 8192) @ Vb(bf16)[4096,1024]
// A tiles via TDM (double-buffered); V tiles transposed by manual scatter.
// ---------------------------------------------------------------------------
static __device__ __forceinline__ void stage_ctx_A_manual(
    bf16_t (*As)[LP], const bf16_t* __restrict__ P, size_t ldP,
    int m0, int k0, int tid)
{
    const int r = tid >> 1, c = (tid & 1) * 32;
    const bf16_t* pa = P + (size_t)(m0 + r) * ldP + k0 + c;
    #pragma unroll
    for (int j = 0; j < 4; ++j)
        *reinterpret_cast<int4*>(&As[r][c + j * 8]) =
            *reinterpret_cast<const int4*>(pa + j * 8);
}

static __device__ __forceinline__ void stage_ctx_B(
    bf16_t (*Bt)[LP], const bf16_t* __restrict__ Vb,
    int n0, int k0, int tid)
{
    const int rB = tid >> 2;           // k 0..63
    const int cB = (tid & 3) * 32;     // n 0..96
    const bf16_t* vb = Vb + (size_t)(k0 + rB) * HID + n0 + cB;
    #pragma unroll
    for (int j = 0; j < 4; ++j) {
        bf16x8 bv = *reinterpret_cast<const bf16x8*>(vb + j * 8);
        #pragma unroll
        for (int e = 0; e < 8; ++e) Bt[cB + j * 8 + e][rB] = bv[e];
    }
}

__global__ __launch_bounds__(256) void context_kernel(
    const bf16_t* __restrict__ P, const bf16_t* __restrict__ Vb,
    float* __restrict__ Out)
{
    __shared__ bf16_t As[2][BM][LP];
    __shared__ bf16_t Bt[2][BN][LP];
    GEMM_WAVE_SETUP();
    const int m0 = blockIdx.x * BM;
    const int n0 = blockIdx.y * BN;
    const size_t ldP = 2 * SEQ;        // 8192 bf16 slots/row (in-place pitch)
    const int NIT = SEQ / BK;

#if defined(HAVE_TDM)
    if (wave == 0) {
        tdm_load_2d(lds_off_of(&As[0][0][0]), P + (size_t)m0 * ldP,
                    (unsigned)ldP, SEQ, BK, BM, (unsigned)ldP);
        __builtin_amdgcn_s_wait_tensorcnt(0);
    }
#else
    stage_ctx_A_manual(As[0], P, ldP, m0, 0, tid);
#endif
    stage_ctx_B(Bt[0], Vb, n0, 0, tid);
    __syncthreads();

    for (int it = 0; it < NIT; ++it) {
        const int cur = it & 1;
        if (it + 1 < NIT) {
            const int kn = (it + 1) * BK;
#if defined(HAVE_TDM)
            if (wave == 0)
                tdm_load_2d(lds_off_of(&As[cur ^ 1][0][0]),
                            P + (size_t)m0 * ldP + kn,
                            (unsigned)ldP, SEQ, BK, BM, (unsigned)ldP);
#else
            stage_ctx_A_manual(As[cur ^ 1], P, ldP, m0, kn, tid);
#endif
            stage_ctx_B(Bt[cur ^ 1], Vb, n0, kn, tid);
        }
        wave_mma(As[cur], Bt[cur], waveM, waveN, lrow, kh, acc);
#if defined(HAVE_TDM)
        if (wave == 0) __builtin_amdgcn_s_wait_tensorcnt(0);
#endif
        __syncthreads();
    }

    #pragma unroll
    for (int mt = 0; mt < 2; ++mt) {
        const int mBase = m0 + waveM * 32 + mt * 16 + kh * 8;
        #pragma unroll
        for (int nt = 0; nt < 4; ++nt) {
            const int n = n0 + waveN * 64 + nt * 16 + lrow;
            #pragma unroll
            for (int i = 0; i < 8; ++i)
                Out[(size_t)(mBase + i) * HID + n] = acc[mt][nt][i];
        }
    }
}

// ---------------------------------------------------------------------------
// Launch: q,k,v,Wq,bq,Wk,bk,Wv,bv -> out [4096,1024] f32
// Workspace (88 MB): Qb[0,8M) Kb[8M,16M) Vb[16M,24M) scores f32 [24M,88M)
// (scores overwritten in place with bf16 softmax probabilities).
// ---------------------------------------------------------------------------
extern "C" void kernel_launch(void* const* d_in, const int* in_sizes, int n_in,
                              void* d_out, int out_size, void* d_ws, size_t ws_size,
                              hipStream_t stream) {
    const float* q  = (const float*)d_in[0];
    const float* k  = (const float*)d_in[1];
    const float* v  = (const float*)d_in[2];
    const float* Wq = (const float*)d_in[3];
    const float* bq = (const float*)d_in[4];
    const float* Wk = (const float*)d_in[5];
    const float* bk = (const float*)d_in[6];
    const float* Wv = (const float*)d_in[7];
    const float* bv = (const float*)d_in[8];

    char* ws = (char*)d_ws;
    bf16_t* Qb = (bf16_t*)(ws + 0);
    bf16_t* Kb = (bf16_t*)(ws + (size_t)8 * 1024 * 1024);
    bf16_t* Vb = (bf16_t*)(ws + (size_t)16 * 1024 * 1024);
    float*  Sc = (float*)(ws + (size_t)24 * 1024 * 1024);

    dim3 blk(256);
    dim3 gproj(SEQ / BM, HID / BN);          // 32 x 8
    proj_kernel<<<gproj, blk, 0, stream>>>(q, Wq, bq, Qb);
    proj_kernel<<<gproj, blk, 0, stream>>>(k, Wk, bk, Kb);
    proj_kernel<<<gproj, blk, 0, stream>>>(v, Wv, bv, Vb);

    dim3 gsc(SEQ / BM, SEQ / BN);            // 32 x 32
    scores_kernel<<<gsc, blk, 0, stream>>>(Qb, Kb, Sc);

    softmax_kernel<<<SEQ, blk, 0, stream>>>(Sc);

    dim3 gctx(SEQ / BM, HID / BN);           // 32 x 8
    context_kernel<<<gctx, blk, 0, stream>>>((const bf16_t*)Sc, Vb, (float*)d_out);
}
